// LSTMLayer_37641093382840
// MI455X (gfx1250) — compile-verified, matrix operand order checked by reading, not run on hardware
//
#include <hip/hip_runtime.h>
#include <hip/hip_bf16.h>

// Problem dims (fixed by the reference)
constexpr int BB  = 32;     // batch
constexpr int SS  = 512;    // sequence
constexpr int INN = 1024;   // input features
constexpr int HH  = 1024;   // hidden
constexpr int G4H = 4 * HH; // stacked gates i,f,g,o

// Recurrent kernel config: 64 persistent workgroups, 4 waves (1 per gate),
// each WG owns a 16-wide hidden slice.
constexpr int NWG = 64;
constexpr int NH  = HH / NWG; // 16

typedef __attribute__((ext_vector_type(16))) __bf16 v16bf;
typedef __attribute__((ext_vector_type(8)))  __bf16 v8bf;
typedef __attribute__((ext_vector_type(8)))  float  v8f;

__device__ __forceinline__ unsigned short f32_to_bf16(float f) {
  unsigned u = __float_as_uint(f);
  u += 0x7FFFu + ((u >> 16) & 1u);   // round-to-nearest-even
  return (unsigned short)(u >> 16);
}

__device__ __forceinline__ v8bf ldv8(const unsigned short* p) {
  return *reinterpret_cast<const v8bf*>(p);
}

// 16-bit WMMA operand fragment: per-lane two contiguous 8-elem chunks,
// second chunk 16 elements further along K (ISA 16-bit A/B layout).
__device__ __forceinline__ v16bf load_frag(const unsigned short* base) {
  v8bf c0 = ldv8(base);
  v8bf c1 = ldv8(base + 16);
  return __builtin_shufflevector(c0, c1, 0,1,2,3,4,5,6,7,8,9,10,11,12,13,14,15);
}

__device__ __forceinline__ float sigmoidf(float x) {
  return 1.0f / (1.0f + __expf(-x));
}

// ---------------------------------------------------------------------------
// Prep: fp32 -> bf16 conversion (grid-stride)
// ---------------------------------------------------------------------------
__global__ void cvt_bf16_kernel(const float* __restrict__ src,
                                unsigned short* __restrict__ dst, int n) {
  int i = blockIdx.x * blockDim.x + threadIdx.x;
  int stride = gridDim.x * blockDim.x;
  for (; i < n; i += stride) dst[i] = f32_to_bf16(src[i]);
}

// h state buffer init: broadcast h0[H] across batch, in bf16
__global__ void init_h_kernel(const float* __restrict__ h0,
                              unsigned short* __restrict__ hbf) {
  int i = blockIdx.x * blockDim.x + threadIdx.x;
  if (i < BB * HH) hbf[i] = f32_to_bf16(h0[i % HH]);
}

// ---------------------------------------------------------------------------
// Phase 1: xg[t][b][col] = x[b,t,:] . Wx[col,:] + bx[col] + bh[col]
// 64x64 output tile per wave (4x4 WMMA subtiles, K=1024):
//   per k-step: 4 A-frags + 4 B-frags (256 B/lane) feed 16 WMMAs
//   -> 32 FLOP per L2 byte, ~3x less L2 traffic than a 32x16 tile.
// ---------------------------------------------------------------------------
__global__ __launch_bounds__(256)
void gemm_xg_kernel(const unsigned short* __restrict__ xbf,   // [B*S, IN] bf16
                    const unsigned short* __restrict__ wxbf,  // [4H, IN] bf16
                    const float* __restrict__ bx,
                    const float* __restrict__ bh,
                    float* __restrict__ xg) {                 // [S, B, 4H] f32
  const int wave  = (blockIdx.x * blockDim.x + threadIdx.x) >> 5;
  const int lane  = threadIdx.x & 31;
  const int tilesN = G4H / 64;              // 64
  const int tileM = wave / tilesN;          // 0..255
  const int tileN = wave % tilesN;
  const int r    = lane & 15;
  const int half = lane >> 4;
  const int m0 = tileM * 64;                // 64 rows stay inside one batch b
  const int n0 = tileN * 64;

  const unsigned short* ap[4];
  const unsigned short* bp[4];
#pragma unroll
  for (int i = 0; i < 4; ++i) {
    ap[i] = xbf  + (size_t)(m0 + i * 16 + r) * INN + half * 8;
    bp[i] = wxbf + (size_t)(n0 + i * 16 + r) * INN + half * 8;
  }

  v8f acc[4][4] = {};
  for (int k0 = 0; k0 < INN; k0 += 32) {
    v16bf av[4], bv[4];
#pragma unroll
    for (int i = 0; i < 4; ++i) {
      av[i] = load_frag(ap[i] + k0);
      bv[i] = load_frag(bp[i] + k0);
    }
#pragma unroll
    for (int mi = 0; mi < 4; ++mi) {
#pragma unroll
      for (int ni = 0; ni < 4; ++ni) {
        acc[mi][ni] = __builtin_amdgcn_wmma_f32_16x16x32_bf16(
            false, av[mi], false, bv[ni], (short)0, acc[mi][ni], false, false);
      }
    }
  }

  const int bidx  = m0 >> 9;      // batch index (m = b*S + t, S=512)
  const int tbase = m0 & 511;
#pragma unroll
  for (int ni = 0; ni < 4; ++ni) {
    int col = n0 + ni * 16 + r;
    float bias = bx[col] + bh[col];
#pragma unroll
    for (int mi = 0; mi < 4; ++mi) {
#pragma unroll
      for (int rr = 0; rr < 8; ++rr) {
        int t = tbase + mi * 16 + half * 8 + rr;  // C layout: M = rr + 8*(lane>=16)
        xg[((size_t)t * BB + bidx) * G4H + col] = acc[mi][ni][rr] + bias;
      }
    }
  }
}

// ---------------------------------------------------------------------------
// Phase 2: persistent recurrent kernel. 64 WGs x 128 threads (4 waves).
// Wave g computes gate g's [32 x 16] pre-activation tile for this WG's
// hidden slice, stages it in LDS; the cell update runs locally; h_t goes to a
// device-visible bf16 buffer; one atomic-counter grid barrier per step.
// ---------------------------------------------------------------------------
__global__ __launch_bounds__(128)
void lstm_recurrent_kernel(const float* __restrict__ xg,        // [S,B,4H]
                           const unsigned short* __restrict__ whbf, // [4H,H] bf16
                           unsigned short* __restrict__ hbf,    // [B,H] bf16 state
                           const float* __restrict__ c0,
                           float* __restrict__ out,             // hs | h_last | c_last
                           unsigned* __restrict__ ctr) {
  __shared__ float gsm[4][32][NH];   // staged gate tiles (i,f,g,o)
  __shared__ float csm[32][NH];      // persistent cell state for this slice

  const int n0   = blockIdx.x * NH;
  const int gate = threadIdx.x >> 5;     // wave id == gate id
  const int lane = threadIdx.x & 31;
  const int r    = lane & 15;
  const int half = lane >> 4;

  for (int e = threadIdx.x; e < 32 * NH; e += blockDim.x) {
    csm[e / NH][e % NH] = c0[n0 + (e % NH)];
  }
  __syncthreads();

  // B fragment base: Wh row (gate*H + n0 + r) is B-column (n0+r), contiguous K
  const unsigned short* bp = whbf + (size_t)(gate * HH + n0 + r) * HH + half * 8;

  for (int t = 0; t < SS; ++t) {
    // ---- gates_tile = h_prev @ Wh[gate-slice]^T (bf16 WMMA, f32 acc) ----
    const unsigned short* a0p = hbf + (size_t)r * HH + half * 8;        // rows 0..15
    const unsigned short* a1p = a0p + (size_t)16 * HH;                  // rows 16..31
    v8f acc0 = {}, acc1 = {};
    for (int k0 = 0; k0 < HH; k0 += 32) {
      v16bf av0 = load_frag(a0p + k0);
      v16bf av1 = load_frag(a1p + k0);
      v16bf bv  = load_frag(bp + k0);
      acc0 = __builtin_amdgcn_wmma_f32_16x16x32_bf16(false, av0, false, bv,
                                                     (short)0, acc0, false, false);
      acc1 = __builtin_amdgcn_wmma_f32_16x16x32_bf16(false, av1, false, bv,
                                                     (short)0, acc1, false, false);
    }

    // ---- add xg (contiguous [B,4H] slice for this t) and stage in LDS ----
    const float* xgs = xg + (size_t)t * BB * G4H + (size_t)gate * HH + n0 + r;
#pragma unroll
    for (int rr = 0; rr < 8; ++rr) {
      int row0 = half * 8 + rr;
      gsm[gate][row0][r]      = acc0[rr] + xgs[(size_t)row0 * G4H];
      gsm[gate][16 + row0][r] = acc1[rr] + xgs[(size_t)(16 + row0) * G4H];
    }
    __syncthreads();

    // ---- LSTM cell: 32*NH elems, processed as 256 horizontal pairs ----
    for (int p = threadIdx.x; p < (32 * NH) / 2; p += blockDim.x) {
      int b = p >> 3;              // NH/2 = 8 pairs per batch row
      int n = (p & 7) * 2;
      float hv[2];
      unsigned short hb[2];
#pragma unroll
      for (int j = 0; j < 2; ++j) {
        int nn = n + j;
        float iv = sigmoidf(gsm[0][b][nn]);
        float fv = sigmoidf(gsm[1][b][nn]);
        float gv = tanhf(gsm[2][b][nn]);
        float ov = sigmoidf(gsm[3][b][nn]);
        float cv = fv * csm[b][nn] + iv * gv;
        csm[b][nn] = cv;
        hv[j] = ov * tanhf(cv);
        hb[j] = f32_to_bf16(hv[j]);
      }
      // hidden_states[b, t, n0+n .. +1]
      float2 h2; h2.x = hv[0]; h2.y = hv[1];
      *reinterpret_cast<float2*>(out + (size_t)b * SS * HH + (size_t)t * HH + n0 + n) = h2;
      // bf16 h state for next step's GEMM (device-visible after fence)
      unsigned packed = (unsigned)hb[0] | ((unsigned)hb[1] << 16);
      *reinterpret_cast<unsigned*>(hbf + (size_t)b * HH + n0 + n) = packed;
      if (t == SS - 1) {
        size_t hoff = (size_t)BB * SS * HH;
        out[hoff + (size_t)b * HH + n0 + n]     = hv[0];
        out[hoff + (size_t)b * HH + n0 + n + 1] = hv[1];
        size_t coff = hoff + (size_t)BB * HH;
        out[coff + (size_t)b * HH + n0 + n]     = csm[b][n];
        out[coff + (size_t)b * HH + n0 + n + 1] = csm[b][n + 1];
      }
    }

    // ---- device-wide barrier: release h_t, acquire everyone's h_t ----
    __threadfence();               // release: flush h stores to device scope
    __syncthreads();
    if (threadIdx.x == 0) {
      atomicAdd(ctr, 1u);
      unsigned target = (unsigned)(t + 1) * (unsigned)NWG;
      while (atomicAdd(ctr, 0u) < target) {
        __builtin_amdgcn_s_sleep(2);
      }
    }
    __syncthreads();
    __threadfence();               // acquire: invalidate WGP caches for h reads
  }
}

// ---------------------------------------------------------------------------
// Launch
// ---------------------------------------------------------------------------
extern "C" void kernel_launch(void* const* d_in, const int* in_sizes, int n_in,
                              void* d_out, int out_size, void* d_ws, size_t ws_size,
                              hipStream_t stream) {
  const float* x  = (const float*)d_in[0];
  const float* h0 = (const float*)d_in[1];
  const float* c0 = (const float*)d_in[2];
  const float* Wx = (const float*)d_in[3];
  const float* bx = (const float*)d_in[4];
  const float* Wh = (const float*)d_in[5];
  const float* bh = (const float*)d_in[6];
  float* out = (float*)d_out;

  // Workspace layout
  char* ws = (char*)d_ws;
  size_t off = 0;
  float* xg = (float*)(ws + off);             off += (size_t)SS * BB * G4H * 4;  // 256 MB
  unsigned short* xbf  = (unsigned short*)(ws + off); off += (size_t)BB * SS * INN * 2; // 32 MB
  unsigned short* wxbf = (unsigned short*)(ws + off); off += (size_t)G4H * INN * 2;     // 8 MB
  unsigned short* whbf = (unsigned short*)(ws + off); off += (size_t)G4H * HH * 2;      // 8 MB
  unsigned short* hbf  = (unsigned short*)(ws + off); off += (size_t)BB * HH * 2;
  unsigned* ctr = (unsigned*)(ws + off);      off += 256;

  hipMemsetAsync(ctr, 0, sizeof(unsigned), stream);

  cvt_bf16_kernel<<<4096, 256, 0, stream>>>(x,  xbf,  BB * SS * INN);
  cvt_bf16_kernel<<<2048, 256, 0, stream>>>(Wx, wxbf, G4H * INN);
  cvt_bf16_kernel<<<2048, 256, 0, stream>>>(Wh, whbf, G4H * HH);
  init_h_kernel<<<(BB * HH + 255) / 256, 256, 0, stream>>>(h0, hbf);

  // 256 M-tiles x 64 N-tiles = 16384 waves, 8 waves/block
  gemm_xg_kernel<<<2048, 256, 0, stream>>>(xbf, wxbf, bx, bh, xg);

  // Persistent recurrent scan: 64 co-resident workgroups
  lstm_recurrent_kernel<<<NWG, 128, 0, stream>>>(xg, whbf, hbf, c0, out, ctr);
}
